// AimNet2Core_69415261438046
// MI455X (gfx1250) — matrix operand order repaired and checked by reading, not compiled, not run on hardware
//
#include <hip/hip_runtime.h>

typedef __attribute__((ext_vector_type(16))) _Float16 v16h;
typedef __attribute__((ext_vector_type(8)))  float    v8f;

// ---------------------------------------------------------------------------
// WMMA fragment index maps (CDNA5 ISA 7.12.2, wave32)
// A (16x32 f16): lane<16 holds M=lane, K = e (e<8) / e+8 (e>=8)
//                lane>=16 holds M=lane-16, K = e+8 (e<8) / e+16 (e>=8)
// B (32x16 f16): lane = N (mod 16); same K map as A (lane>=16 adds 8)
// C/D (16x16 f32): vgpr r -> (M = r + 8*(lane>=16), N = lane&15)
// ---------------------------------------------------------------------------
__device__ __forceinline__ int bmap(int lane, int e) {
  return (e & 7) + ((e & 8) ? 16 : 0) + ((lane & 16) ? 8 : 0);
}

__device__ __forceinline__ int afrag_idx(int m, int k) {
  int kt = k >> 5, kk = k & 31;
  int grp = (kk >> 3) & 1;                 // kk in [8,16) or [24,32)
  int lane = m + (grp ? 16 : 0);
  int e = (kk & 7) + ((kk >= 16) ? 8 : 0);
  return (kt * 32 + lane) * 16 + e;
}

__device__ __forceinline__ float gelu_f(float x) {
  const float c = 0.7978845608028654f;
  return 0.5f * x * (1.0f + tanhf(c * (x + 0.044715f * x * x * x)));
}

// ---------------------------------------------------------------------------
struct ModuleParams {
  const _Float16 *s1, *s2, *a3, *a4, *q3, *q4;   // pre-swizzled f16 B-frags
  const float *s1_b, *s2_b, *a3_b, *a4_b, *q3_b, *q4_b;
  int kt1;                                        // K-tiles of layer 1
};

// ---------------------------------------------------------------------------
__global__ void zero_kernel(float* __restrict__ p, long n) {
  long i = (long)blockIdx.x * blockDim.x + threadIdx.x;
  if (i < n) __builtin_nontemporal_store(0.0f, &p[i]);
}

// Pre-swizzle a [din,dout] fp32 weight into f16 B-fragment order:
// out[((kt*ntiles + nt)*32 + lane)*16 + e] = W[kt*32 + bmap(lane,e)][nt*16 + lane%16]
__global__ void swizzle_kernel(const float* __restrict__ W, _Float16* __restrict__ out,
                               int din, int dout, int ktiles, int ntiles) {
  int i = blockIdx.x * blockDim.x + threadIdx.x;
  int total = ktiles * ntiles * 512;
  if (i >= total) return;
  int e    = i & 15;
  int lane = (i >> 4) & 31;
  int rest = i >> 9;
  int nt = rest % ntiles;
  int kt = rest / ntiles;
  int k  = kt * 32 + bmap(lane, e);
  int nn = nt * 16 + (lane & 15);
  float v = (k < din && nn < dout) ? W[(long)k * dout + nn] : 0.0f;
  out[i] = (_Float16)v;
}

// ---------------------------------------------------------------------------
// Pair channel: 64 threads per pair (feature-per-lane), 4 pairs per block.
// fc (409.6 MB/pass, 3 passes) and u are pure streams -> nontemporal loads so
// the L2-resident reuse buffers (emb gathers, radial/q atomics) stay cached.
// radial[j,f]   += fc[p,f]*emb[j,f] ; vecc[j,d]   += u[p,d]*sum_f(fc*emb_j)
// radial_q[j,f] += fc[p,f]*q[j]     ; vecc_q[j,d] += u[p,d]*q_j*sum_f(fc)
// ---------------------------------------------------------------------------
__global__ void pair_kernel(const float* __restrict__ fc, const float* __restrict__ u,
                            const int* __restrict__ idxj,
                            const float* __restrict__ emb, const float* __restrict__ q,
                            float* __restrict__ radial, float* __restrict__ vecc,
                            float* __restrict__ radial_q, float* __restrict__ vecc_q,
                            int P, int use_q) {
  int group = threadIdx.x >> 6;        // 4 pair-groups per 256-thread block
  int t     = threadIdx.x & 63;        // feature id
  int p     = blockIdx.x * 4 + group;
  bool active = (p < P);

  int j = 0; float fcv = 0.f, ev = 0.f, qj = 0.f;
  if (active) {
    j   = idxj[p];
    fcv = __builtin_nontemporal_load(&fc[(long)p * 64 + t]);   // stream, NT
    ev  = emb[(long)j * 64 + t];                               // reused, RT
    if (use_q) qj = q[j];
  }
  float proto = fcv * ev;
  if (active) {
    atomicAdd(&radial[(long)j * 64 + t], proto);
    if (use_q) atomicAdd(&radial_q[(long)j * 64 + t], fcv * qj);
  }
  // reduce proto and fc over the 64 lanes of this group (2 waves of 32)
  float sp = proto, sf = fcv;
  #pragma unroll
  for (int off = 16; off > 0; off >>= 1) {
    sp += __shfl_down(sp, off);
    sf += __shfl_down(sf, off);
  }
  __shared__ float redp[8], redf[8];
  int wv = threadIdx.x >> 5;
  if ((threadIdx.x & 31) == 0) { redp[wv] = sp; redf[wv] = sf; }
  __syncthreads();
  if (active && t < 3) {
    float s  = redp[2 * group] + redp[2 * group + 1];
    float uv = __builtin_nontemporal_load(&u[(long)p * 3 + t]);  // stream, NT
    atomicAdd(&vecc[(long)j * 3 + t], uv * s);
    if (use_q) {
      float sfc = redf[2 * group] + redf[2 * group + 1];
      atomicAdd(&vecc_q[(long)j * 3 + t], uv * sfc * qj);
    }
  }
}

// ---------------------------------------------------------------------------
// WMMA helpers
// ---------------------------------------------------------------------------
__device__ __forceinline__ v8f tile_mm(const _Float16* A, const _Float16* W,
                                       int ktc, int nt, int ntiles, float bval, int lane) {
  v8f acc;
  #pragma unroll
  for (int r = 0; r < 8; ++r) acc[r] = bval;
  for (int kt = 0; kt < ktc; ++kt) {
    v16h a = *(const v16h*)(A + ((long)kt * 32 + lane) * 16);
    v16h b = *(const v16h*)(W + (((long)kt * ntiles + nt) * 32 + lane) * 16);
    acc = __builtin_amdgcn_wmma_f32_16x16x32_f16(false, a, false, b, (short)0, acc,
                                                 false, false);
  }
  return acc;
}

__device__ __forceinline__ void store_act(v8f acc, _Float16* out, int nt, int lane) {
  int nn  = lane & 15;
  int mhi = (lane >> 4) & 1;
  #pragma unroll
  for (int r = 0; r < 8; ++r) {
    float v = gelu_f(acc[r]);
    out[afrag_idx(r + mhi * 8, nt * 16 + nn)] = (_Float16)v;
  }
}

// ---------------------------------------------------------------------------
// MLP: one wave per 16-atom tile. msg build -> L1 -> L2 -> {a3,a4 | q3,q4}
// ---------------------------------------------------------------------------
__global__ __launch_bounds__(32) void mlp_kernel(
    const float* __restrict__ radial,   const float* __restrict__ vecc,
    const float* __restrict__ radial_q, const float* __restrict__ vecc_q,
    const float* __restrict__ emb_in,   const float* __restrict__ q_in,
    float* __restrict__ emb_out,        float* __restrict__ q_out,
    ModuleParams mp, int n, int use_q) {
  __shared__ __attribute__((aligned(32))) _Float16 bufA[5 * 512];  // msg / h2
  __shared__ __attribute__((aligned(32))) _Float16 bufB[4 * 512];  // h1
  __shared__ __attribute__((aligned(32))) _Float16 bufC[1 * 512];  // a3 act
  __shared__ __attribute__((aligned(32))) _Float16 bufD[1 * 512];  // q3 act

  int lane = threadIdx.x;
  int base = blockIdx.x * 16;
  int din_pad = mp.kt1 * 32;

  // build msg = [radial(64) | vnorm(1) | radial_q(64) | vnorm_q(1) | 0-pad]
  for (int idx = lane; idx < 16 * din_pad; idx += 32) {
    int m = idx / din_pad, k = idx - m * din_pad;
    int atom = base + m;
    float val = 0.0f;
    if (atom < n) {
      if (k < 64) val = radial[(long)atom * 64 + k];
      else if (k == 64) {
        float x = vecc[atom * 3], y = vecc[atom * 3 + 1], z = vecc[atom * 3 + 2];
        val = sqrtf(x * x + y * y + z * z + 1e-12f);
      } else if (use_q) {
        if (k < 129) val = radial_q[(long)atom * 64 + (k - 65)];
        else if (k == 129) {
          float x = vecc_q[atom * 3], y = vecc_q[atom * 3 + 1], z = vecc_q[atom * 3 + 2];
          val = sqrtf(x * x + y * y + z * z + 1e-12f);
        }
      }
    }
    bufA[afrag_idx(m, k)] = (_Float16)val;
  }
  __syncthreads();

  int nn  = lane & 15;
  int mhi = (lane >> 4) & 1;

  // L1: [16,din] x [din,128] -> bufB (gelu)
  for (int nt = 0; nt < 8; ++nt)
    store_act(tile_mm(bufA, mp.s1, mp.kt1, nt, 8, mp.s1_b[nt * 16 + nn], lane), bufB, nt, lane);
  __syncthreads();
  // L2: [16,128] x [128,64] -> bufA (gelu)
  for (int nt = 0; nt < 4; ++nt)
    store_act(tile_mm(bufB, mp.s2, 4, nt, 4, mp.s2_b[nt * 16 + nn], lane), bufA, nt, lane);
  __syncthreads();
  // a3 / q3: [16,64] x [64,32] -> bufC / bufD (gelu)
  for (int nt = 0; nt < 2; ++nt)
    store_act(tile_mm(bufA, mp.a3, 2, nt, 2, mp.a3_b[nt * 16 + nn], lane), bufC, nt, lane);
  for (int nt = 0; nt < 2; ++nt)
    store_act(tile_mm(bufA, mp.q3, 2, nt, 2, mp.q3_b[nt * 16 + nn], lane), bufD, nt, lane);
  __syncthreads();
  // a4: [16,32] x [32,64] -> emb_out = emb_in + da  (write-once -> NT store)
  for (int nt = 0; nt < 4; ++nt) {
    v8f acc = tile_mm(bufC, mp.a4, 1, nt, 4, mp.a4_b[nt * 16 + nn], lane);
    int col = nt * 16 + nn;
    #pragma unroll
    for (int r = 0; r < 8; ++r) {
      int atom = base + r + mhi * 8;
      if (atom < n) {
        float v = emb_in[(long)atom * 64 + col] + acc[r];
        __builtin_nontemporal_store(v, &emb_out[(long)atom * 64 + col]);
      }
    }
  }
  // q4: [16,32] x [32,1] -> q_out = q_in + dq  (only N==0 column valid)
  {
    v8f acc = tile_mm(bufD, mp.q4, 1, 0, 1, (nn == 0) ? mp.q4_b[0] : 0.0f, lane);
    if (nn == 0) {
      #pragma unroll
      for (int r = 0; r < 8; ++r) {
        int atom = base + r + mhi * 8;
        if (atom < n) q_out[atom] = q_in[atom] + acc[r];
      }
    }
  }
}

// ---------------------------------------------------------------------------
__global__ void count_kernel(const int* __restrict__ sys, float* __restrict__ cnt, int n) {
  int i = blockIdx.x * blockDim.x + threadIdx.x;
  if (i < n) atomicAdd(&cnt[sys[i]], 1.0f);
}
__global__ void syssum_kernel(const float* __restrict__ q, const int* __restrict__ sys,
                              float* __restrict__ qs, int n) {
  int i = blockIdx.x * blockDim.x + threadIdx.x;
  if (i < n) atomicAdd(&qs[sys[i]], q[i]);
}
__global__ void corr_kernel(float* __restrict__ q, const int* __restrict__ sys,
                            const float* __restrict__ qs, const float* __restrict__ cnt,
                            const float* __restrict__ tot, int n) {
  int i = blockIdx.x * blockDim.x + threadIdx.x;
  if (i < n) {
    int s = sys[i];
    q[i] += (tot[s] - qs[s]) / fmaxf(cnt[s], 1.0f);
  }
}
__global__ void pack_kernel(const float* __restrict__ emb, const float* __restrict__ q,
                            float* __restrict__ out, int n) {
  long total = (long)n * 65;
  long i = (long)blockIdx.x * blockDim.x + threadIdx.x;
  if (i < total) {
    long a = i / 65; int f = (int)(i - a * 65);
    float v = (f < 64) ? emb[a * 64 + f] : q[a];
    __builtin_nontemporal_store(v, &out[i]);
  }
}

// ---------------------------------------------------------------------------
extern "C" void kernel_launch(void* const* d_in, const int* in_sizes, int n_in,
                              void* d_out, int out_size, void* d_ws, size_t ws_size,
                              hipStream_t stream) {
  const float* emb0 = (const float*)d_in[0];
  const float* fc   = (const float*)d_in[1];
  const float* u    = (const float*)d_in[2];
  const int*   pidx = (const int*)d_in[3];
  const int*   sysi = (const int*)d_in[4];
  const float* tot  = (const float*)d_in[5];
  int N = in_sizes[0] / 64;
  int P = in_sizes[1] / 64;
  int S = in_sizes[5];
  const int* idxj = pidx + P;   // pair_indices row 1

  // bump allocator on workspace
  char* wp = (char*)d_ws;
  auto carve = [&](size_t bytes) -> void* {
    void* r = (void*)wp;
    wp += (bytes + 255) & ~(size_t)255;
    return r;
  };
  float* radial    = (float*)carve((size_t)N * 64 * 4);
  float* radial_q  = (float*)carve((size_t)N * 64 * 4);
  float* vecc      = (float*)carve((size_t)N * 3 * 4);
  float* vecc_q    = (float*)carve((size_t)N * 3 * 4);
  float* embbuf[2] = {(float*)carve((size_t)N * 64 * 4), (float*)carve((size_t)N * 64 * 4)};
  float* qbuf[2]   = {(float*)carve((size_t)N * 4),      (float*)carve((size_t)N * 4)};
  float* qs  = (float*)carve((size_t)S * 4);
  float* cnt = (float*)carve((size_t)S * 4);

  auto zero = [&](float* p, long n) {
    zero_kernel<<<dim3((unsigned)((n + 255) / 256)), dim3(256), 0, stream>>>(p, n);
  };
  auto swz = [&](const float* W, int din, int dout) -> _Float16* {
    int kt = (din + 31) / 32, ntl = (dout + 15) / 16;
    int total = kt * ntl * 512;
    _Float16* o = (_Float16*)carve((size_t)total * 2);
    swizzle_kernel<<<dim3((unsigned)((total + 255) / 256)), dim3(256), 0, stream>>>(
        W, o, din, dout, kt, ntl);
    return o;
  };

  // params flattened in setup_inputs() dict insertion order:
  // per module: s1_w, s1_b, s2_w, s2_b, a3_w, a3_b, a4_w, a4_b, q3_w, q3_b, q4_w, q4_b
  ModuleParams mp[3];
  for (int m = 0; m < 3; ++m) {
    int b = 6 + m * 12;
    int din1 = in_sizes[b] / 128;
    mp[m].s1   = swz((const float*)d_in[b + 0], din1, 128);
    mp[m].s1_b = (const float*)d_in[b + 1];
    mp[m].s2   = swz((const float*)d_in[b + 2], 128, 64);
    mp[m].s2_b = (const float*)d_in[b + 3];
    mp[m].a3   = swz((const float*)d_in[b + 4], 64, 32);
    mp[m].a3_b = (const float*)d_in[b + 5];
    mp[m].a4   = swz((const float*)d_in[b + 6], 32, 64);
    mp[m].a4_b = (const float*)d_in[b + 7];
    mp[m].q3   = swz((const float*)d_in[b + 8], 64, 32);
    mp[m].q3_b = (const float*)d_in[b + 9];
    mp[m].q4   = swz((const float*)d_in[b + 10], 32, 1);
    mp[m].q4_b = (const float*)d_in[b + 11];
    mp[m].kt1  = (din1 + 31) / 32;
  }

  // per-system atom counts (recomputed every call: deterministic)
  zero(cnt, S);
  count_kernel<<<dim3((unsigned)((N + 255) / 256)), dim3(256), 0, stream>>>(sysi, cnt, N);
  zero(qbuf[0], N);

  const float* emb_cur = emb0;
  const float* q_cur   = qbuf[0];
  for (int m = 0; m < 3; ++m) {
    float* emb_next = embbuf[m & 1];
    float* q_next   = qbuf[(m + 1) & 1];
    int use_q = (m > 0) ? 1 : 0;

    zero(radial, (long)N * 64);
    zero(vecc, (long)N * 3);
    if (use_q) { zero(radial_q, (long)N * 64); zero(vecc_q, (long)N * 3); }

    pair_kernel<<<dim3((unsigned)((P + 3) / 4)), dim3(256), 0, stream>>>(
        fc, u, idxj, emb_cur, q_cur, radial, vecc, radial_q, vecc_q, P, use_q);

    mlp_kernel<<<dim3((unsigned)((N + 15) / 16)), dim3(32), 0, stream>>>(
        radial, vecc, radial_q, vecc_q, emb_cur, q_cur, emb_next, q_next,
        mp[m], N, use_q);

    zero(qs, S);
    syssum_kernel<<<dim3((unsigned)((N + 255) / 256)), dim3(256), 0, stream>>>(q_next, sysi, qs, N);
    corr_kernel<<<dim3((unsigned)((N + 255) / 256)), dim3(256), 0, stream>>>(
        q_next, sysi, qs, cnt, tot, N);

    emb_cur = emb_next;
    q_cur   = q_next;
  }

  long total_out = (long)N * 65;
  pack_kernel<<<dim3((unsigned)((total_out + 255) / 256)), dim3(256), 0, stream>>>(
      emb_cur, q_cur, (float*)d_out, N);
}
